// SparseAttention_25683904430351
// MI455X (gfx1250) — matrix operand order repaired and checked by reading, not compiled, not run on hardware
//
#include <hip/hip_runtime.h>
#include <hip/hip_bf16.h>

// MI455X / gfx1250, wave32. Multi-head attention forward:
//   cvt(weights f32->f16) -> gemm_qkv x3 (WMMA) -> flash-attn (WMMA + async LDS
//   double-buffering via global_load_async_to_lds_b128) -> gemm_out (WMMA)

typedef __attribute__((ext_vector_type(16))) _Float16 v16h;
typedef __attribute__((ext_vector_type(8)))  _Float16 v8h;
typedef __attribute__((ext_vector_type(8)))  float    v8f;

#define B_  8
#define S_  512
#define E_  1024
#define H_  16
#define DH_ 64

__device__ __forceinline__ v8f wmma_f16(v16h a, v16h b, v8f c) {
  // (neg_a, A, neg_b, B, c_mod, C, reuse_a, reuse_b)
  return __builtin_amdgcn_wmma_f32_16x16x32_f16(false, a, false, b, (short)0, c,
                                                false, false);
}

// Async global->LDS copy, 16B per lane, tracked by ASYNCcnt (CDNA5).
// lds_off = workgroup-relative LDS byte offset (low 32 bits of generic pointer).
__device__ __forceinline__ void async_g2l_b128(const void* lds_ptr, const void* gptr) {
  unsigned dst = (unsigned)(size_t)lds_ptr;
  unsigned long long src = (unsigned long long)(size_t)gptr;
  asm volatile("global_load_async_to_lds_b128 %0, %1, off"
               :: "v"(dst), "v"(src) : "memory");
}
#define WAIT_ASYNCCNT(n) asm volatile("s_wait_asynccnt " #n ::: "memory")

// ---- fragment loaders (wave32 WMMA; contiguous k order, same for A and B) ----------------------------
// A (16x32 f16): lane L<16 holds row L, K = kk+0..15;
//                lane L>=16 holds row L-16, K = kk+16..31.
__device__ __forceinline__ v16h load_a_f16(const _Float16* base, int ld, int m0, int kk) {
  int ln = threadIdx.x & 31, hf = ln >> 4, row = m0 + (ln & 15);
  const _Float16* p = base + row * ld + kk + hf * 16;
  v8h lo = *(const v8h*)p;
  v8h hi = *(const v8h*)(p + 8);
  v16h r;
#pragma unroll
  for (int i = 0; i < 8; ++i) { r[i] = lo[i]; r[i + 8] = hi[i]; }
  return r;
}

// Same A fragment but converting from an f32 row-major source on the fly.
__device__ __forceinline__ v16h load_a_f32(const float* base, int ld, int m0, int kk) {
  int ln = threadIdx.x & 31, hf = ln >> 4, row = m0 + (ln & 15);
  const float* p = base + row * ld + kk + hf * 16;
  v16h r;
#pragma unroll
  for (int i = 0; i < 8; ++i) { r[i] = (_Float16)p[i]; r[i + 8] = (_Float16)p[8 + i]; }
  return r;
}

// B (32x16 f16) for NT gemm: B(k,n) = base[(n0+n)*ld + k].
// lane L<16 holds col L, K = kk+0..15; lane L>=16 holds col L-16, K = kk+16..31.
__device__ __forceinline__ v16h load_bT_f16(const _Float16* base, int ld, int n0, int kk) {
  int ln = threadIdx.x & 31, hf = ln >> 4;
  const _Float16* p = base + (n0 + (ln & 15)) * ld + kk + hf * 16;
  v8h lo = *(const v8h*)p;
  v8h hi = *(const v8h*)(p + 8);
  v16h r;
#pragma unroll
  for (int i = 0; i < 8; ++i) { r[i] = lo[i]; r[i + 8] = hi[i]; }
  return r;
}

// B (32x16 f16) for NN gemm: B(k,n) = base[(kk+k)*ld + n0+n] (column gather).
__device__ __forceinline__ v16h load_bN_f16(const _Float16* base, int ld, int n0, int kk) {
  int ln = threadIdx.x & 31, hf = ln >> 4, col = n0 + (ln & 15);
  v16h r;
#pragma unroll
  for (int i = 0; i < 16; ++i) r[i] = base[(kk + hf * 16 + i) * ld + col];
  return r;
}

__device__ __forceinline__ float half_max(float v) {   // reduce over 16-lane half
  v = fmaxf(v, __shfl_xor(v, 1, 32));
  v = fmaxf(v, __shfl_xor(v, 2, 32));
  v = fmaxf(v, __shfl_xor(v, 4, 32));
  v = fmaxf(v, __shfl_xor(v, 8, 32));
  return v;
}
__device__ __forceinline__ float half_sum(float v) {
  v += __shfl_xor(v, 1, 32);
  v += __shfl_xor(v, 2, 32);
  v += __shfl_xor(v, 4, 32);
  v += __shfl_xor(v, 8, 32);
  return v;
}

// ---- kernels -----------------------------------------------------------------

__global__ void cvt_f32_to_f16(const float* __restrict__ src,
                               _Float16* __restrict__ dst, int n) {
  int i = blockIdx.x * blockDim.x + threadIdx.x;
  int stride = gridDim.x * blockDim.x;
  for (; i < n; i += stride) dst[i] = (_Float16)src[i];
}

// C[m,n] = X[m,:]·W[n,:] + bias[n] + pos[s,n]; scatter into [B,H,S,Dh] f16.
// Block = 8 waves, block tile 128(M)x64(N), wave tile 16x64. Grid = 512.
__global__ void __launch_bounds__(256) gemm_qkv_kernel(
    const float* __restrict__ X, const _Float16* __restrict__ W16,
    const float* __restrict__ bias, const float* __restrict__ pos,
    _Float16* __restrict__ dst) {
  int bm = blockIdx.x >> 4, bn = blockIdx.x & 15;
  int wave = threadIdx.x >> 5;
  int m0 = bm * 128 + wave * 16;
  int n0 = bn * 64;
  v8f acc[4] = {};
  for (int kk = 0; kk < E_; kk += 32) {
    v16h a  = load_a_f32(X, E_, m0, kk);
    v16h b0 = load_bT_f16(W16, E_, n0 +  0, kk);
    v16h b1 = load_bT_f16(W16, E_, n0 + 16, kk);
    v16h b2 = load_bT_f16(W16, E_, n0 + 32, kk);
    v16h b3 = load_bT_f16(W16, E_, n0 + 48, kk);
    acc[0] = wmma_f16(a, b0, acc[0]);
    acc[1] = wmma_f16(a, b1, acc[1]);
    acc[2] = wmma_f16(a, b2, acc[2]);
    acc[3] = wmma_f16(a, b3, acc[3]);
  }
  int ln = threadIdx.x & 31, hf = ln >> 4;
#pragma unroll
  for (int nb = 0; nb < 4; ++nb) {
    int n = n0 + nb * 16 + (ln & 15);
    float bv = bias[n];
    int hh = n >> 6, dd = n & 63;
#pragma unroll
    for (int j = 0; j < 8; ++j) {
      int m = m0 + j + 8 * hf;
      int bb = m >> 9, ss = m & (S_ - 1);
      float val = acc[nb][j] + bv + pos[ss * E_ + n];
      dst[(size_t)(((bb * H_) + hh) * S_ + ss) * DH_ + dd] = (_Float16)val;
    }
  }
}

// Flash attention: one block = 8 waves = 8 q-tiles of the same (b,h).
// Double-buffered K/V chunks (32 keys) in LDS via async global->LDS copies.
// Grid = B*H*4 = 512.
__global__ void __launch_bounds__(256) attn_kernel(
    const _Float16* __restrict__ Q16, const _Float16* __restrict__ K16,
    const _Float16* __restrict__ V16, _Float16* __restrict__ WV16) {
  __shared__ _Float16 Kt[2][32 * 72];     // 32 keys x 64 d (pad 8), double-buffered
  __shared__ _Float16 Vt[2][32 * 72];
  __shared__ _Float16 Pt[8 * 16 * 48];    // per-wave 16x32 prob tile (pad to 48)

  int bh = blockIdx.x >> 2, qb = blockIdx.x & 3;
  int b = bh >> 4, h = bh & 15;
  int wave = threadIdx.x >> 5;
  int ln = threadIdx.x & 31, hf = ln >> 4;
  int q0 = (qb * 8 + wave) * 16;

  const _Float16* Qb = Q16 + (size_t)(b * H_ + h) * S_ * DH_;
  const _Float16* Kb = K16 + (size_t)(b * H_ + h) * S_ * DH_;
  const _Float16* Vb = V16 + (size_t)(b * H_ + h) * S_ * DH_;
  _Float16* PtW = Pt + wave * 16 * 48;

  v16h qa0 = load_a_f16(Qb, DH_, q0, 0);
  v16h qa1 = load_a_f16(Qb, DH_, q0, 32);

  v8f o[4] = {};
  float m_run[8], l_run[8];
#pragma unroll
  for (int j = 0; j < 8; ++j) { m_run[j] = -1e30f; l_run[j] = 0.0f; }

  int crow = threadIdx.x >> 3;          // 0..31
  int ccol = (threadIdx.x & 7) * 8;     // 0..56 (16B per lane)

  // prologue: async-prefetch chunk 0 into buffer 0 (2 ops/wave outstanding)
  async_g2l_b128(&Kt[0][crow * 72 + ccol], Kb + (size_t)crow * DH_ + ccol);
  async_g2l_b128(&Vt[0][crow * 72 + ccol], Vb + (size_t)crow * DH_ + ccol);

  int buf = 0;
  for (int kk = 0; kk < S_; kk += 32) {
    __syncthreads();  // all waves finished reading buf^1 before it is overwritten
    if (kk + 32 < S_) {   // block-uniform: prefetch next chunk into other buffer
      async_g2l_b128(&Kt[buf ^ 1][crow * 72 + ccol],
                     Kb + (size_t)(kk + 32 + crow) * DH_ + ccol);
      async_g2l_b128(&Vt[buf ^ 1][crow * 72 + ccol],
                     Vb + (size_t)(kk + 32 + crow) * DH_ + ccol);
      WAIT_ASYNCCNT(0x2);  // async loads complete in order: current chunk landed
    } else {
      WAIT_ASYNCCNT(0x0);
    }
    __syncthreads();  // every wave's async data visible to every other wave

    const _Float16* Kc = Kt[buf];
    const _Float16* Vc = Vt[buf];

    // scores: 16 q x 32 keys, contracting over Dh=64 (two k-steps per key chunk)
    v16h kb00 = load_bT_f16(Kc, 72, 0, 0);
    v16h kb01 = load_bT_f16(Kc, 72, 0, 32);
    v16h kb10 = load_bT_f16(Kc, 72, 16, 0);
    v16h kb11 = load_bT_f16(Kc, 72, 16, 32);
    v8f s0 = {}, s1 = {};
    s0 = wmma_f16(qa0, kb00, s0);
    s0 = wmma_f16(qa1, kb01, s0);
    s1 = wmma_f16(qa0, kb10, s1);
    s1 = wmma_f16(qa1, kb11, s1);

    // online softmax; element j of C-frag lives at row q0 + j + 8*hf
#pragma unroll
    for (int j = 0; j < 8; ++j) {
      float a0 = s0[j] * 0.125f;   // 1/sqrt(Dh)
      float a1 = s1[j] * 0.125f;
      float mn = fmaxf(m_run[j], half_max(fmaxf(a0, a1)));
      float alpha = __expf(m_run[j] - mn);
      float p0 = __expf(a0 - mn);
      float p1 = __expf(a1 - mn);
      l_run[j] = l_run[j] * alpha + half_sum(p0 + p1);
      m_run[j] = mn;
#pragma unroll
      for (int nb = 0; nb < 4; ++nb) o[nb][j] *= alpha;
      int prow = j + 8 * hf;
      PtW[prow * 48 + (ln & 15)]      = (_Float16)p0;
      PtW[prow * 48 + 16 + (ln & 15)] = (_Float16)p1;
    }
    __syncthreads();  // Pt visible across lanes (uniform trip count in block)

    // O += P(16x32) x V(32x64)
    v16h pa = load_a_f16(PtW, 48, 0, 0);
#pragma unroll
    for (int nb = 0; nb < 4; ++nb)
      o[nb] = wmma_f16(pa, load_bN_f16(Vc, 72, nb * 16, 0), o[nb]);

    buf ^= 1;
  }

  // normalize and write [B,S,E] f16 (head-interleaved back to embedding dim)
#pragma unroll
  for (int j = 0; j < 8; ++j) {
    float inv = 1.0f / l_run[j];
    int q = q0 + j + 8 * hf;
#pragma unroll
    for (int nb = 0; nb < 4; ++nb) {
      int d = nb * 16 + (ln & 15);
      WV16[((size_t)(b * S_) + q) * E_ + h * DH_ + d] = (_Float16)(o[nb][j] * inv);
    }
  }
}

// out[m,n] = X16[m,:]·Wo[n,:] + bo[n], f32 output. Same tiling as gemm_qkv.
__global__ void __launch_bounds__(256) gemm_out_kernel(
    const _Float16* __restrict__ X16, const _Float16* __restrict__ Wo16,
    const float* __restrict__ bias, float* __restrict__ out) {
  int bm = blockIdx.x >> 4, bn = blockIdx.x & 15;
  int wave = threadIdx.x >> 5;
  int m0 = bm * 128 + wave * 16;
  int n0 = bn * 64;
  v8f acc[4] = {};
  for (int kk = 0; kk < E_; kk += 32) {
    v16h a  = load_a_f16(X16, E_, m0, kk);
    v16h b0 = load_bT_f16(Wo16, E_, n0 +  0, kk);
    v16h b1 = load_bT_f16(Wo16, E_, n0 + 16, kk);
    v16h b2 = load_bT_f16(Wo16, E_, n0 + 32, kk);
    v16h b3 = load_bT_f16(Wo16, E_, n0 + 48, kk);
    acc[0] = wmma_f16(a, b0, acc[0]);
    acc[1] = wmma_f16(a, b1, acc[1]);
    acc[2] = wmma_f16(a, b2, acc[2]);
    acc[3] = wmma_f16(a, b3, acc[3]);
  }
  int ln = threadIdx.x & 31, hf = ln >> 4;
#pragma unroll
  for (int nb = 0; nb < 4; ++nb) {
    int n = n0 + nb * 16 + (ln & 15);
    float bv = bias[n];
#pragma unroll
    for (int j = 0; j < 8; ++j) {
      int m = m0 + j + 8 * hf;
      out[(size_t)m * E_ + n] = acc[nb][j] + bv;
    }
  }
}

// ---- host launch -------------------------------------------------------------
extern "C" void kernel_launch(void* const* d_in, const int* in_sizes, int n_in,
                              void* d_out, int out_size, void* d_ws, size_t ws_size,
                              hipStream_t stream) {
  (void)in_sizes; (void)n_in; (void)out_size; (void)ws_size;
  const float* q   = (const float*)d_in[0];
  const float* k   = (const float*)d_in[1];
  const float* v   = (const float*)d_in[2];
  const float* Wq  = (const float*)d_in[3];
  const float* bq  = (const float*)d_in[4];
  const float* Wk  = (const float*)d_in[5];
  const float* bk  = (const float*)d_in[6];
  const float* Wv  = (const float*)d_in[7];
  const float* bv  = (const float*)d_in[8];
  const float* Wo  = (const float*)d_in[9];
  const float* bo  = (const float*)d_in[10];
  const float* pos = (const float*)d_in[11];
  float* out = (float*)d_out;

  // workspace layout (f16 elements): 4x 1M weights, 3x 4M QKV, 1x 4M attn out = 40 MB
  _Float16* ws   = (_Float16*)d_ws;
  _Float16* wq16 = ws;
  _Float16* wk16 = wq16 + (size_t)E_ * E_;
  _Float16* wv16 = wk16 + (size_t)E_ * E_;
  _Float16* wo16 = wv16 + (size_t)E_ * E_;
  _Float16* Q16  = wo16 + (size_t)E_ * E_;                 // [B,H,S,Dh]
  _Float16* K16  = Q16 + (size_t)B_ * S_ * E_;
  _Float16* V16  = K16 + (size_t)B_ * S_ * E_;
  _Float16* WV16 = V16 + (size_t)B_ * S_ * E_;             // [B*S, E]

  const int nW = E_ * E_;
  cvt_f32_to_f16<<<1024, 256, 0, stream>>>(Wq, wq16, nW);
  cvt_f32_to_f16<<<1024, 256, 0, stream>>>(Wk, wk16, nW);
  cvt_f32_to_f16<<<1024, 256, 0, stream>>>(Wv, wv16, nW);
  cvt_f32_to_f16<<<1024, 256, 0, stream>>>(Wo, wo16, nW);

  gemm_qkv_kernel<<<512, 256, 0, stream>>>(q, wq16, bq, pos, Q16);
  gemm_qkv_kernel<<<512, 256, 0, stream>>>(k, wk16, bk, pos, K16);
  gemm_qkv_kernel<<<512, 256, 0, stream>>>(v, wv16, bv, pos, V16);

  attn_kernel<<<512, 256, 0, stream>>>(Q16, K16, V16, WV16);

  gemm_out_kernel<<<512, 256, 0, stream>>>(WV16, wo16, bo, out);
}
